// Yolov4Head_36266703847636
// MI455X (gfx1250) — compile-verified, matrix operand order checked by reading, not run on hardware
//
#include <hip/hip_runtime.h>
#include <hip/hip_bf16.h>

// ---------------- constants from the reference ----------------
#define IMG_W     76
#define IMG_H     76
#define HW        (IMG_W * IMG_H)      // 5776
#define NANCH     3
#define NCLS      80
#define NCH       255                  // 3*(5+80)
#define NCAND     (HW * NANCH)         // 17328
#define TOPK      300
#define TOPK_PAD  320                  // 20 tiles of 16
#define NWORDS    10                   // 320 bits
#define SCALE_XY  1.2f
#define OBJ_TH    0.5f
#define SCORE_TH  0.05f
#define NMS_TH    0.7f

typedef __attribute__((ext_vector_type(16))) _Float16 v16h;
typedef __attribute__((ext_vector_type(8)))  float    v8f;

__device__ __forceinline__ float sigmoidf_fast(float x) {
    return 1.0f / (1.0f + __expf(-x));
}

// ---------------------------------------------------------------------------
// Kernel 1: fused decode. One thread per (b, hw); loops anchors + channels.
// All 255 channel loads are coalesced across lanes (lane -> consecutive hw).
// ---------------------------------------------------------------------------
__global__ __launch_bounds__(256) void decode_kernel(
    const float* __restrict__ x, const float* __restrict__ anchors,
    float* __restrict__ wsScores,   // [B][NCAND] masked score (-1 if invalid)
    float* __restrict__ wsBoxes,    // [B][NCAND][4] clipped xyxy
    int*   __restrict__ wsLabels,   // [B][NCAND]
    int B)
{
    int gid = blockIdx.x * 256 + threadIdx.x;
    if (gid >= B * HW) return;
    int b  = gid / HW;
    int hw = gid - b * HW;
    int h  = hw / IMG_W;
    int w  = hw - h * IMG_W;

    const float* xb = x + (size_t)b * NCH * HW + hw;

    float aw[NANCH], ah[NANCH];
    #pragma unroll
    for (int a = 0; a < NANCH; ++a) { aw[a] = anchors[2 * a]; ah[a] = anchors[2 * a + 1]; }

    #pragma unroll
    for (int a = 0; a < NANCH; ++a) {
        const float* xa = xb + (size_t)a * 85 * HW;
        // gfx1250 speculative prefetch of the next anchor's first cachelines
        if (a + 1 < NANCH) __builtin_prefetch(xa + (size_t)85 * HW, 0, 0);

        float tx = xa[0];
        float ty = xa[(size_t)1 * HW];
        float tw = xa[(size_t)2 * HW];
        float th = xa[(size_t)3 * HW];
        float to = xa[(size_t)4 * HW];

        // argmax over raw logits (sigmoid is monotonic)
        float mc = xa[(size_t)5 * HW];
        int lab = 0;
        for (int c = 1; c < NCLS; ++c) {
            float v = xa[(size_t)(5 + c) * HW];
            if (v > mc) { mc = v; lab = c; }
        }

        float sx = SCALE_XY * sigmoidf_fast(tx) - 0.5f * (SCALE_XY - 1.0f);
        float sy = SCALE_XY * sigmoidf_fast(ty) - 0.5f * (SCALE_XY - 1.0f);
        float bx = (sx + (float)w) * (1.0f / IMG_W);
        float by = (sy + (float)h) * (1.0f / IMG_H);
        float bw = fminf(fmaxf(__expf(tw) * aw[a], 0.0f), 2.0f);
        float bh = fminf(fmaxf(__expf(th) * ah[a], 0.0f), 2.0f);
        float x1 = bx - 0.5f * bw, y1 = by - 0.5f * bh;
        float x2 = x1 + bw,        y2 = y1 + bh;
        x1 = fminf(fmaxf(x1, 0.0f), 1.0f); y1 = fminf(fmaxf(y1, 0.0f), 1.0f);
        x2 = fminf(fmaxf(x2, 0.0f), 1.0f); y2 = fminf(fmaxf(y2, 0.0f), 1.0f);

        float obj   = sigmoidf_fast(to);
        float score = sigmoidf_fast(mc) * obj;
        bool  valid = (obj >= OBJ_TH) && (score >= SCORE_TH);

        int n = hw * NANCH + a;                    // (h, w, a) flatten order
        size_t base = (size_t)b * NCAND + n;
        wsScores[base] = valid ? score : -1.0f;
        float* o = wsBoxes + base * 4;
        o[0] = x1; o[1] = y1; o[2] = x2; o[3] = y2;
        wsLabels[base] = lab;
    }
}

// ---------------------------------------------------------------------------
// Kernel 2: per-batch top-300 (iterative argmax, value desc / index asc ties).
// Scores live in registers (17/thread, fully unrolled). One block per batch.
// ---------------------------------------------------------------------------
#define TK_THREADS 1024
#define TK_PER_THR 17   // ceil(17328/1024)

__global__ __launch_bounds__(TK_THREADS) void topk_kernel(
    const float* __restrict__ wsScores, const float* __restrict__ wsBoxes,
    const int* __restrict__ wsLabels,
    float* __restrict__ selBoxes,   // [B][TOPK_PAD][4]
    float* __restrict__ selArea,    // [B][TOPK_PAD]
    int*   __restrict__ selValid,   // [B][TOPK_PAD]
    float* __restrict__ outBoxes,   // [B][TOPK][4]
    float* __restrict__ outScores,  // [B][TOPK]
    float* __restrict__ outLabels)  // [B][TOPK]
{
    int b = blockIdx.x, tid = threadIdx.x;
    const float* sb = wsScores + (size_t)b * NCAND;

    float regs[TK_PER_THR];
    #pragma unroll
    for (int k = 0; k < TK_PER_THR; ++k) {
        int n = tid + (k << 10);
        regs[k] = (n < NCAND) ? sb[n] : -2e30f;
    }

    __shared__ float rv[TK_THREADS];
    __shared__ int   ri[TK_THREADS];
    __shared__ int   selIdx[TOPK];
    __shared__ float selVal[TOPK];

    for (int r = 0; r < TOPK; ++r) {
        float bv = -3e30f; int bi = 0x7fffffff;
        #pragma unroll
        for (int k = 0; k < TK_PER_THR; ++k) {
            if (regs[k] > bv) { bv = regs[k]; bi = tid + (k << 10); }
        }
        rv[tid] = bv; ri[tid] = bi;
        __syncthreads();
        for (int s = TK_THREADS / 2; s > 0; s >>= 1) {
            if (tid < s) {
                float ov = rv[tid + s]; int oi = ri[tid + s];
                if (ov > rv[tid] || (ov == rv[tid] && oi < ri[tid])) { rv[tid] = ov; ri[tid] = oi; }
            }
            __syncthreads();
        }
        int win = ri[0];
        if (tid == 0) { selIdx[r] = win; selVal[r] = rv[0]; }
        #pragma unroll
        for (int k = 0; k < TK_PER_THR; ++k)        // constant-index removal, stays in VGPRs
            if (win == tid + (k << 10)) regs[k] = -2e30f;
        __syncthreads();
    }

    for (int t = tid; t < TOPK_PAD; t += TK_THREADS) {
        float* s4 = selBoxes + ((size_t)b * TOPK_PAD + t) * 4;
        if (t < TOPK) {
            int idx = selIdx[t]; float v = selVal[t];
            bool valid = (v >= 0.0f);
            const float* bx = wsBoxes + ((size_t)b * NCAND + idx) * 4;
            float x1 = bx[0], y1 = bx[1], x2 = bx[2], y2 = bx[3];
            s4[0] = x1; s4[1] = y1; s4[2] = x2; s4[3] = y2;
            selArea[b * TOPK_PAD + t]  = (x2 - x1) * (y2 - y1);
            selValid[b * TOPK_PAD + t] = valid ? 1 : 0;
            float* o4 = outBoxes + ((size_t)b * TOPK + t) * 4;
            o4[0] = x1; o4[1] = y1; o4[2] = x2; o4[3] = y2;
            outScores[b * TOPK + t] = valid ? v : 0.0f;
            outLabels[b * TOPK + t] = (float)wsLabels[(size_t)b * NCAND + idx];
        } else {
            s4[0] = s4[1] = s4[2] = s4[3] = 0.0f;
            selArea[b * TOPK_PAD + t] = 0.0f;
            selValid[b * TOPK_PAD + t] = 0;
        }
    }
}

// ---------------------------------------------------------------------------
// Kernel 3: suppression bitmask. grid=(B, 20 row-tiles), block=512.
// Wave 0 computes the 16x320 pairwise areaSum stripe with WMMA
// (rank-2 update [a_i,1]x[1,a_j]^T embedded in K-slots 0/1 of f16 WMMA),
// then 512 threads evaluate IoU and set bits.
// ---------------------------------------------------------------------------
__global__ __launch_bounds__(512) void nms_pairs_kernel(
    const float* __restrict__ selBoxes, const float* __restrict__ selArea,
    unsigned* __restrict__ supp)   // [B][TOPK_PAD][NWORDS]
{
    __shared__ float    sAreaSum[16 * TOPK_PAD];   // 20 KB
    __shared__ float    sBox[TOPK_PAD * 4];        // 5 KB
    __shared__ unsigned sBits[16 * NWORDS];

    int b = blockIdx.x, it = blockIdx.y;
    int i0 = it * 16;

    for (int t = threadIdx.x; t < TOPK_PAD * 4; t += blockDim.x)
        sBox[t] = selBoxes[(size_t)b * TOPK_PAD * 4 + t];
    for (int t = threadIdx.x; t < 16 * NWORDS; t += blockDim.x)
        sBits[t] = 0u;
    __syncthreads();

    if (threadIdx.x < 32) {        // wave 0, EXEC all ones
        int lane = threadIdx.x;
        int m = lane & 15;
        float ai = selArea[b * TOPK_PAD + i0 + m];
        v16h va = {};
        if (lane < 16) { va[0] = (_Float16)ai; va[1] = (_Float16)1.0f; }   // A[m][0]=a_i, A[m][1]=1
        for (int jt = 0; jt < TOPK_PAD / 16; ++jt) {
            float aj = selArea[b * TOPK_PAD + jt * 16 + m];
            v16h vb = {};
            if (lane < 16) { vb[0] = (_Float16)1.0f; vb[1] = (_Float16)aj; } // B[0][n]=1, B[1][n]=a_j
            v8f c = {};
            c = __builtin_amdgcn_wmma_f32_16x16x32_f16(
                    false, va, false, vb, (short)0, c, false, false);
            int n = lane & 15;
            int mbase = (lane < 16) ? 0 : 8;
            #pragma unroll
            for (int r = 0; r < 8; ++r)
                sAreaSum[(mbase + r) * TOPK_PAD + jt * 16 + n] = c[r];
        }
    }
    __syncthreads();

    for (int p = threadIdx.x; p < 16 * TOPK_PAD; p += blockDim.x) {
        int ii = p / TOPK_PAD;
        int j  = p - ii * TOPK_PAD;
        int gi = i0 + ii;
        if (gi >= TOPK || j >= TOPK || j <= gi) continue;
        const float* bi = &sBox[gi * 4];
        const float* bj = &sBox[j * 4];
        float lx = fmaxf(bi[0], bj[0]);
        float ly = fmaxf(bi[1], bj[1]);
        float rx = fminf(bi[2], bj[2]);
        float ry = fminf(bi[3], bj[3]);
        float iw = fmaxf(rx - lx, 0.0f);
        float ih = fmaxf(ry - ly, 0.0f);
        float inter = iw * ih;
        float uni = sAreaSum[ii * TOPK_PAD + j] - inter + 1e-9f;
        if (inter > NMS_TH * uni)
            atomicOr(&sBits[ii * NWORDS + (j >> 5)], 1u << (j & 31));
    }
    __syncthreads();

    for (int t = threadIdx.x; t < 16 * NWORDS; t += blockDim.x)
        supp[((size_t)b * TOPK_PAD + i0) * NWORDS + t] = sBits[t];
}

// ---------------------------------------------------------------------------
// Kernel 4: sequential keep propagation (matches fori_loop exactly). 1 wave.
// ---------------------------------------------------------------------------
__global__ __launch_bounds__(32) void propagate_kernel(
    const unsigned* __restrict__ supp, const int* __restrict__ selValid,
    float* __restrict__ outKeep)
{
    int b = blockIdx.x, lane = threadIdx.x;
    __shared__ unsigned kw[NWORDS];
    if (lane < NWORDS) {
        unsigned v = 0;
        for (int k = 0; k < 32; ++k) {
            int i = lane * 32 + k;
            if (i < TOPK && selValid[b * TOPK_PAD + i]) v |= 1u << k;
        }
        kw[lane] = v;
    }
    __syncthreads();
    const unsigned* srow = supp + (size_t)b * TOPK_PAD * NWORDS;
    for (int i = 0; i < TOPK; ++i) {
        bool alive = (kw[i >> 5] >> (i & 31)) & 1u;
        __syncthreads();
        if (alive && lane < NWORDS) kw[lane] &= ~srow[(size_t)i * NWORDS + lane];
        __syncthreads();
    }
    for (int i = lane; i < TOPK; i += 32)
        outKeep[b * TOPK + i] = ((kw[i >> 5] >> (i & 31)) & 1u) ? 1.0f : 0.0f;
}

// ---------------------------------------------------------------------------
extern "C" void kernel_launch(void* const* d_in, const int* in_sizes, int n_in,
                              void* d_out, int out_size, void* d_ws, size_t ws_size,
                              hipStream_t stream)
{
    const float* x       = (const float*)d_in[0];
    const float* anchors = (const float*)d_in[1];
    int B = in_sizes[0] / (NCH * HW);   // 32

    char* ws = (char*)d_ws;
    size_t off = 0;
    auto alloc = [&](size_t bytes) -> void* {
        void* p = (void*)(ws + off);
        off += (bytes + 255) & ~(size_t)255;
        return p;
    };
    float*    wsScores = (float*)alloc((size_t)B * NCAND * sizeof(float));
    float*    wsBoxes  = (float*)alloc((size_t)B * NCAND * 4 * sizeof(float));
    int*      wsLabels = (int*)  alloc((size_t)B * NCAND * sizeof(int));
    float*    selBoxes = (float*)alloc((size_t)B * TOPK_PAD * 4 * sizeof(float));
    float*    selArea  = (float*)alloc((size_t)B * TOPK_PAD * sizeof(float));
    int*      selValid = (int*)  alloc((size_t)B * TOPK_PAD * sizeof(int));
    unsigned* supp     = (unsigned*)alloc((size_t)B * TOPK_PAD * NWORDS * sizeof(unsigned));

    float* outBoxes  = (float*)d_out;             // [B][300][4]
    float* outScores = outBoxes + (size_t)B * TOPK * 4;
    float* outLabels = outScores + (size_t)B * TOPK;
    float* outKeep   = outLabels + (size_t)B * TOPK;

    int total = B * HW;
    decode_kernel<<<(total + 255) / 256, 256, 0, stream>>>(
        x, anchors, wsScores, wsBoxes, wsLabels, B);

    topk_kernel<<<B, TK_THREADS, 0, stream>>>(
        wsScores, wsBoxes, wsLabels, selBoxes, selArea, selValid,
        outBoxes, outScores, outLabels);

    dim3 g(B, TOPK_PAD / 16);
    nms_pairs_kernel<<<g, 512, 0, stream>>>(selBoxes, selArea, supp);

    propagate_kernel<<<B, 32, 0, stream>>>(supp, selValid, outKeep);
}
